// MultiTaskDrugNN_47691316855323
// MI455X (gfx1250) — matrix operand order repaired and checked by reading, not compiled, not run on hardware
//
#include <hip/hip_runtime.h>

typedef __bf16 bf16_t;
typedef __attribute__((ext_vector_type(16))) __bf16 v16bf;
typedef __attribute__((ext_vector_type(8)))  __bf16 v8bf;
typedef __attribute__((ext_vector_type(8)))  float  v8f;

#define B_SZ  8192
#define IN_SZ 19000
#define S_SZ  256
#define PD_SZ 128
#define P_SZ  16
#define D_SZ  64

#define TM 64
#define TN 64
#define TK 32

// ---------------------------------------------------------------------------
// Kernel 1: h = relu(x @ W1 + b1)   [8192 x 19000] @ [19000 x 256]
// bf16 WMMA (v_wmma_f32_16x16x32_bf16), f32 accumulate.
// Block: 128 threads (4 waves). Tile: 64(M) x 64(N), K-step 32.
// Wave w computes C[0:64, w*16:(w+1)*16] as 4 accumulators of 16x16.
// ---------------------------------------------------------------------------
__global__ __launch_bounds__(128) void gemm_relu_bf16_kernel(
    const float* __restrict__ x,    // [B_SZ, IN_SZ]
    const float* __restrict__ W1,   // [IN_SZ, S_SZ]
    const float* __restrict__ b1,   // [S_SZ]
    float* __restrict__ h)          // [B_SZ, S_SZ]
{
    __shared__ __attribute__((aligned(16))) bf16_t As[TM][TK];  // m-major
    __shared__ __attribute__((aligned(16))) bf16_t Bs[TN][TK];  // n-major (transposed)

    const int t    = threadIdx.x;
    const int lane = t & 31;
    const int wave = t >> 5;        // 0..3
    const int col  = lane & 15;     // N (and M for A-frag) index within 16
    const int half = lane >> 4;     // lane half selects K sub-range
    const int m0   = blockIdx.x * TM;
    const int n0   = blockIdx.y * TN;

    v8f acc[4];
    #pragma unroll
    for (int i = 0; i < 4; ++i)
        acc[i] = v8f{0.f, 0.f, 0.f, 0.f, 0.f, 0.f, 0.f, 0.f};

    const int nIter = (IN_SZ + TK - 1) / TK;  // 594 (last step zero-padded)

    for (int kt = 0; kt < nIter; ++kt) {
        const int k0 = kt * TK;
        __syncthreads();

        // --- stage x tile -> As (bf16). 64 rows x 32 cols = 512 float4 groups.
        #pragma unroll
        for (int i = 0; i < 4; ++i) {
            const int g   = t + i * 128;       // 0..511
            const int row = g >> 3;            // 0..63
            const int cg  = g & 7;             // float4 group within row
            const int kk  = k0 + cg * 4;
            const float* src = x + (size_t)(m0 + row) * IN_SZ + kk;
            float4 v = make_float4(0.f, 0.f, 0.f, 0.f);
            if (kk + 3 < IN_SZ) {
                v = *(const float4*)src;
                if (kt + 1 < nIter)
                    __builtin_prefetch(src + TK, 0, 1);   // global_prefetch next K-tile
            } else {
                if (kk + 0 < IN_SZ) v.x = src[0];
                if (kk + 1 < IN_SZ) v.y = src[1];
                if (kk + 2 < IN_SZ) v.z = src[2];
                if (kk + 3 < IN_SZ) v.w = src[3];
            }
            As[row][cg * 4 + 0] = (bf16_t)v.x;
            As[row][cg * 4 + 1] = (bf16_t)v.y;
            As[row][cg * 4 + 2] = (bf16_t)v.z;
            As[row][cg * 4 + 3] = (bf16_t)v.w;
        }

        // --- stage W1 tile -> Bs transposed (n-major). 32 k-rows x 64 n-cols.
        #pragma unroll
        for (int i = 0; i < 4; ++i) {
            const int g  = t + i * 128;        // 0..511
            const int kl = g >> 4;             // 0..31
            const int ng = g & 15;             // float4 group within n
            const int kk = k0 + kl;
            float4 v = make_float4(0.f, 0.f, 0.f, 0.f);
            if (kk < IN_SZ)
                v = *(const float4*)(W1 + (size_t)kk * S_SZ + n0 + ng * 4);
            Bs[ng * 4 + 0][kl] = (bf16_t)v.x;
            Bs[ng * 4 + 1][kl] = (bf16_t)v.y;
            Bs[ng * 4 + 2][kl] = (bf16_t)v.z;
            Bs[ng * 4 + 3][kl] = (bf16_t)v.w;
        }

        __syncthreads();

        // --- B fragment: lane holds column (wave*16+col), K = half*8 .. +7, half*8+16 .. +23
        v16bf bfrag;
        {
            const v8bf* bp0 = (const v8bf*)&Bs[wave * 16 + col][half * 8];
            v8bf blo = bp0[0];
            v8bf bhi = bp0[2];
            #pragma unroll
            for (int j = 0; j < 8; ++j) { bfrag[j] = blo[j]; bfrag[8 + j] = bhi[j]; }
        }

        // --- 4 A fragments (M sub-tiles) + WMMA
        #pragma unroll
        for (int mt = 0; mt < 4; ++mt) {
            const int row = mt * 16 + col;
            // ISA 16-bit A layout: V0-3 = K[half*8 .. +7], V4-7 = K[half*8+16 .. +23]
            const v8bf* ap0 = (const v8bf*)&As[row][half * 8];
            const v8bf* ap1 = (const v8bf*)&As[row][half * 8 + 16];
            v8bf alo = ap0[0];
            v8bf ahi = ap1[0];
            v16bf afrag;
            #pragma unroll
            for (int j = 0; j < 8; ++j) { afrag[j] = alo[j]; afrag[8 + j] = ahi[j]; }

            acc[mt] = __builtin_amdgcn_wmma_f32_16x16x32_bf16(
                /*neg_a=*/false, afrag, /*neg_b=*/false, bfrag,
                /*c_mod=*/(short)0, acc[mt],
                /*reuse_a=*/false, /*reuse_b=*/false);
        }
    }

    // --- epilogue: +bias, ReLU, store. C layout: lane half picks M 0-7 / 8-15.
    const int   n_g  = n0 + wave * 16 + col;
    const float bias = b1[n_g];
    #pragma unroll
    for (int mt = 0; mt < 4; ++mt) {
        #pragma unroll
        for (int v = 0; v < 8; ++v) {
            const int m_g = m0 + mt * 16 + half * 8 + v;
            const float r = fmaxf(acc[mt][v] + bias, 0.f);
            h[(size_t)m_g * S_SZ + n_g] = r;
        }
    }
}

// ---------------------------------------------------------------------------
// Kernel 2: per-sample pathway expert + drug head (fused).
// One block (128 threads) per sample; thread t owns PD dim t.
// ---------------------------------------------------------------------------
__global__ __launch_bounds__(128) void pathway_head_kernel(
    const float* __restrict__ h,          // [B_SZ, S_SZ]
    const int*   __restrict__ drug_idx,   // [B_SZ]
    const float* __restrict__ Wp,         // [P_SZ, S_SZ, PD_SZ]
    const float* __restrict__ bp,         // [P_SZ, PD_SZ]
    const float* __restrict__ Wd,         // [D_SZ, PD_SZ]
    const float* __restrict__ bd,         // [D_SZ]
    const int*   __restrict__ drug_to_pw, // [D_SZ]
    float*       __restrict__ out)        // [B_SZ]
{
    __shared__ float hs[S_SZ];
    __shared__ float red[PD_SZ];

    const int b    = blockIdx.x;
    const int t    = threadIdx.x;
    const int drug = drug_idx[b];
    const int pw   = drug_to_pw[drug];

    hs[t]       = h[(size_t)b * S_SZ + t];
    hs[t + 128] = h[(size_t)b * S_SZ + t + 128];
    __syncthreads();

    float acc = bp[pw * PD_SZ + t];
    const float* wp = Wp + ((size_t)pw * S_SZ) * PD_SZ + t;
    #pragma unroll 8
    for (int s = 0; s < S_SZ; ++s)
        acc = fmaf(hs[s], wp[(size_t)s * PD_SZ], acc);

    const float v = fmaxf(acc, 0.f);
    red[t] = v * Wd[drug * PD_SZ + t];
    __syncthreads();

    #pragma unroll
    for (int off = 64; off > 0; off >>= 1) {
        if (t < off) red[t] += red[t + off];
        __syncthreads();
    }
    if (t == 0) out[b] = red[0] + bd[drug];
}

// ---------------------------------------------------------------------------
extern "C" void kernel_launch(void* const* d_in, const int* in_sizes, int n_in,
                              void* d_out, int out_size, void* d_ws, size_t ws_size,
                              hipStream_t stream) {
    (void)in_sizes; (void)n_in; (void)out_size; (void)ws_size;

    const float* x    = (const float*)d_in[0];
    const int*   drug = (const int*)  d_in[1];
    const float* W1   = (const float*)d_in[2];
    const float* b1   = (const float*)d_in[3];
    const float* Wp   = (const float*)d_in[4];
    const float* bp   = (const float*)d_in[5];
    const float* Wd   = (const float*)d_in[6];
    const float* bd   = (const float*)d_in[7];
    const int*   d2p  = (const int*)  d_in[8];
    float*       out  = (float*)d_out;
    float*       h    = (float*)d_ws;   // [B_SZ, S_SZ] f32 = 8 MB scratch

    dim3 g1(B_SZ / TM, S_SZ / TN);      // (128, 4)
    gemm_relu_bf16_kernel<<<g1, 128, 0, stream>>>(x, W1, b1, h);
    pathway_head_kernel<<<B_SZ, 128, 0, stream>>>(h, drug, Wp, bp, Wd, bd, d2p, out);
}
